// HyperRopeViT_11081015623916
// MI455X (gfx1250) — matrix operand order repaired and checked by reading, not compiled, not run on hardware
//
#include <hip/hip_runtime.h>
#include <hip/hip_bf16.h>
#include <math.h>

typedef __attribute__((ext_vector_type(16))) _Float16 v16h;
typedef __attribute__((ext_vector_type(8)))  float    v8f;
typedef __attribute__((ext_vector_type(4)))  _Float16 h4;

#define DEPTH  4
#define BATCH  128
#define NTOK   225
#define PSZ    15
#define DIM    256
#define HEADS  8
#define DH     32
#define MLPD   1024
#define NCLS   13
#define SPECT  200
#define MROWS  (BATCH*NTOK)   // 28800

// ---------------------------------------------------------------------------
// CDNA5 WMMA + async-LDS helpers (gfx1250, wave32)
// ---------------------------------------------------------------------------
__device__ inline v8f wmma16(v16h a, v16h b, v8f c) {
  // (neg_a, A, neg_b, B, c_mod, C, reuse_a, reuse_b)
  return __builtin_amdgcn_wmma_f32_16x16x32_f16(false, a, false, b, (short)0, c,
                                                false, false);
}

union AFrag { v16h v; _Float16 h[16]; };

// A-matrix fragment (16x32): lane L -> row rowBase+(L&15), kb=(L>>4)*8;
// VGPR i<4 holds K = kb+2i, kb+2i+1 ; VGPR i>=4 holds K = 16+kb+...
__device__ inline v16h load_fragA(const _Float16* base, int rowBase, int colBase,
                                  int stride) {
  int lane = threadIdx.x & 31;
  const _Float16* p = base + (size_t)(rowBase + (lane & 15)) * stride + colBase +
                      ((lane >> 4) << 3);
  AFrag f;
#pragma unroll
  for (int i = 0; i < 4; ++i) {
    f.h[2*i]     = p[2*i];
    f.h[2*i+1]   = p[2*i+1];
    f.h[8+2*i]   = p[16+2*i];
    f.h[8+2*i+1] = p[16+2*i+1];
  }
  return f.v;
}

// B-matrix fragment (32x16) from (n,k) LDS storage:
// lane L -> col n = rowBase+(L&15); kb=(L>>4)*8; elems 0-7 K=kb..kb+7, 8-15 K=16+kb...
__device__ inline v16h load_fragB(const _Float16* base, int rowBase, int colBase,
                                  int stride) {
  int lane = threadIdx.x & 31;
  const _Float16* p = base + (size_t)(rowBase + (lane & 15)) * stride + colBase +
                      ((lane >> 4) << 3);
  AFrag f;
#pragma unroll
  for (int i = 0; i < 16; ++i) f.h[i] = p[i < 8 ? i : i + 8];
  return f.v;
}

// gfx1250 async DMA: copy 16 bytes global -> LDS, tracked by ASYNCcnt.
// LDS aperture places the byte offset in addr[31:0], so truncating a generic
// shared pointer yields the LDS address the instruction needs.
__device__ inline void async_copy16(void* lds, const _Float16* g) {
  unsigned l = (unsigned)(size_t)lds;
  unsigned long long a = (unsigned long long)(size_t)g;
  asm volatile("global_load_async_to_lds_b128 %0, %1, off"
               :: "v"(l), "v"(a) : "memory");
}
__device__ inline void wait_async0() {
  asm volatile("s_wait_asynccnt 0" ::: "memory");
}

// ---------------------------------------------------------------------------
// WMMA GEMM, f16 A (M,K) x f16 B-transposed (N,K) -> f32 or f16 C (M,N).
// Block tile 128x128, BK=32, 8 waves each 32x64 (8 WMMA / wave / K-step).
// Double-buffered LDS filled with global_load_async_to_lds_b128.
// Compile-time variants: ACT (0 none, 1 exact GELU), BIAS, RESID, OUTF16.
// ---------------------------------------------------------------------------
#define BM 128
#define BN 128
#define BK 32

__device__ inline void stage_tile(const _Float16* __restrict__ A,
                                  const _Float16* __restrict__ Bw,
                                  _Float16* AsBuf, _Float16* BsBuf,
                                  int bm, int bn, int K, int k0, int tid) {
#pragma unroll
  for (int it = 0; it < 2; ++it) {
    int e = tid + it * 256;            // 512 16B chunks per tile
    int r = e >> 2;
    int part = (e & 3) * 8;
    async_copy16(&AsBuf[r * BK + part], A  + (size_t)(bm + r) * K + k0 + part);
    async_copy16(&BsBuf[r * BK + part], Bw + (size_t)(bn + r) * K + k0 + part);
  }
}

template <int ACT, bool HAS_BIAS, bool HAS_RESID, bool OUTF16>
__global__ __launch_bounds__(256) void gemm_wmma_kernel(
    const _Float16* __restrict__ A, const _Float16* __restrict__ Bw,
    float* __restrict__ Cf, _Float16* __restrict__ Ch,
    const float* __restrict__ bias, const float* __restrict__ resid,
    int M, int N, int K)
{
  __shared__ _Float16 As[2][BM * BK];
  __shared__ _Float16 Bs[2][BN * BK];
  const int tid  = threadIdx.x;
  const int wid  = tid >> 5;
  const int lane = tid & 31;
  const int bm = blockIdx.y * BM;
  const int bn = blockIdx.x * BN;
  const int wm = (wid & 3) * 32;
  const int wn = (wid >> 2) * 64;

  v8f acc[2][4];
#pragma unroll
  for (int i = 0; i < 2; ++i)
#pragma unroll
    for (int j = 0; j < 4; ++j)
#pragma unroll
      for (int r = 0; r < 8; ++r) acc[i][j][r] = 0.0f;

  const int KT = K / BK;
  stage_tile(A, Bw, As[0], Bs[0], bm, bn, K, 0, tid);
  wait_async0();
  __syncthreads();

  for (int kt = 0; kt < KT; ++kt) {
    int cur = kt & 1;
    if (kt + 1 < KT)   // stream next tile while computing on current one
      stage_tile(A, Bw, As[cur ^ 1], Bs[cur ^ 1], bm, bn, K, (kt + 1) * BK, tid);

    v16h a0 = load_fragA(As[cur], wm,      0, BK);
    v16h a1 = load_fragA(As[cur], wm + 16, 0, BK);
    v16h b0 = load_fragB(Bs[cur], wn,      0, BK);
    v16h b1 = load_fragB(Bs[cur], wn + 16, 0, BK);
    v16h b2 = load_fragB(Bs[cur], wn + 32, 0, BK);
    v16h b3 = load_fragB(Bs[cur], wn + 48, 0, BK);
    acc[0][0] = wmma16(a0, b0, acc[0][0]);
    acc[0][1] = wmma16(a0, b1, acc[0][1]);
    acc[0][2] = wmma16(a0, b2, acc[0][2]);
    acc[0][3] = wmma16(a0, b3, acc[0][3]);
    acc[1][0] = wmma16(a1, b0, acc[1][0]);
    acc[1][1] = wmma16(a1, b1, acc[1][1]);
    acc[1][2] = wmma16(a1, b2, acc[1][2]);
    acc[1][3] = wmma16(a1, b3, acc[1][3]);

    if (kt + 1 < KT) {
      wait_async0();
      __syncthreads();
    }
  }

  // Epilogue: C/D layout -> lane holds col n = lane&15, rows r + 8*(lane>>4).
  const int cn = lane & 15;
  const int rb = (lane >> 4) * 8;
#pragma unroll
  for (int sm = 0; sm < 2; ++sm) {
#pragma unroll
    for (int sn = 0; sn < 4; ++sn) {
      int gn = bn + wn + sn * 16 + cn;
      float bv = HAS_BIAS ? bias[gn] : 0.0f;
#pragma unroll
      for (int r = 0; r < 8; ++r) {
        int gm = bm + wm + sm * 16 + rb + r;
        float v = acc[sm][sn][r] + bv;
        if (ACT == 1) v = 0.5f * v * (1.0f + erff(v * 0.70710678118654752f));
        size_t ci = (size_t)gm * N + gn;
        if (OUTF16) {
          Ch[ci] = (_Float16)v;
        } else {
          if (HAS_RESID) v += resid[ci];
          Cf[ci] = v;
        }
      }
    }
  }
}

// ---------------------------------------------------------------------------
// Fused flash-attention per (b, h): softmax(Q Kt / sqrt(DH)) V via WMMA.
// f16 qkv input (B, N, 3*DIM); f16 output obuf (B, N, DIM), head at cols h*32.
// Tokens padded 225 -> 256, masked in online softmax.
// ---------------------------------------------------------------------------
#define ANP 256

__global__ __launch_bounds__(256) void attn_wmma_kernel(
    const _Float16* __restrict__ qkv, _Float16* __restrict__ obuf)
{
  __shared__ _Float16 Qs[ANP * DH];     // (token, d)
  __shared__ _Float16 Ks[ANP * DH];     // (token, d)
  __shared__ _Float16 Vt[DH * ANP];     // (d, token)
  __shared__ _Float16 Ps[8 * 16 * DH];  // per-wave P transpose scratch
  const int bh = blockIdx.x;
  const int b = bh >> 3, h = bh & 7;
  const int tid = threadIdx.x, wid = tid >> 5, lane = tid & 31;
  const size_t qoff = (size_t)b * NTOK * (3 * DIM) + h * DH;

  for (int e = tid; e < ANP * 4; e += 256) {   // 8-half (16B) chunks
    int n = e >> 2, d0 = (e & 3) * 8;
    uint4 q = {0, 0, 0, 0}, k = q, v = q;
    if (n < NTOK) {
      const _Float16* p = qkv + qoff + (size_t)n * (3 * DIM) + d0;
      q = *(const uint4*)(p);
      k = *(const uint4*)(p + DIM);
      v = *(const uint4*)(p + 2 * DIM);
    }
    *(uint4*)&Qs[n * DH + d0] = q;
    *(uint4*)&Ks[n * DH + d0] = k;
    const _Float16* vh = (const _Float16*)&v;
#pragma unroll
    for (int j = 0; j < 8; ++j) Vt[(d0 + j) * ANP + n] = vh[j];
  }
  __syncthreads();

  const float scale = 0.1767766952966369f;   // DH^-0.5
  const int cn = lane & 15;
  const int rb = (lane >> 4) * 8;
  _Float16* myP = Ps + wid * 16 * DH;

  for (int t = wid; t < 15; t += 8) {
    int qbase = t * 16;
    v16h qf = load_fragA(Qs, qbase, 0, DH);
    v8f o0, o1;
    float mrow[8], lrow[8];
#pragma unroll
    for (int r = 0; r < 8; ++r) { o0[r] = 0.f; o1[r] = 0.f; mrow[r] = -1e30f; lrow[r] = 0.f; }

    for (int j = 0; j < ANP; j += 32) {
      v16h kf0 = load_fragB(Ks, j,      0, DH);
      v16h kf1 = load_fragB(Ks, j + 16, 0, DH);
      v8f s0, s1;
#pragma unroll
      for (int r = 0; r < 8; ++r) { s0[r] = 0.f; s1[r] = 0.f; }
      s0 = wmma16(qf, kf0, s0);
      s1 = wmma16(qf, kf1, s1);

      int c0 = j + cn, c1 = j + 16 + cn;
      float p0[8], p1[8], nm[8];
#pragma unroll
      for (int r = 0; r < 8; ++r) {
        float a0 = (c0 < NTOK) ? s0[r] * scale : -1e30f;
        float a1 = (c1 < NTOK) ? s1[r] * scale : -1e30f;
        p0[r] = a0; p1[r] = a1;
        float mx = fmaxf(a0, a1);
#pragma unroll
        for (int mk = 1; mk < 16; mk <<= 1) mx = fmaxf(mx, __shfl_xor(mx, mk, 32));
        nm[r] = fmaxf(mrow[r], mx);
      }
#pragma unroll
      for (int r = 0; r < 8; ++r) {
        float corr = __expf(mrow[r] - nm[r]);
        lrow[r] *= corr; o0[r] *= corr; o1[r] *= corr;
        float e0 = __expf(p0[r] - nm[r]);
        float e1 = __expf(p1[r] - nm[r]);
        float sm = e0 + e1;
#pragma unroll
        for (int mk = 1; mk < 16; mk <<= 1) sm += __shfl_xor(sm, mk, 32);
        lrow[r] += sm;
        mrow[r] = nm[r];
        int prow = rb + r;             // C-layout -> A-layout via LDS transpose
        myP[prow * DH + cn]      = (_Float16)e0;
        myP[prow * DH + 16 + cn] = (_Float16)e1;
      }
      v16h pf  = load_fragA(myP, 0, 0, DH);      // same-wave DS is in-order
      v16h vb0 = load_fragB(Vt, 0,  j, ANP);
      v16h vb1 = load_fragB(Vt, 16, j, ANP);
      o0 = wmma16(pf, vb0, o0);
      o1 = wmma16(pf, vb1, o1);
    }
#pragma unroll
    for (int r = 0; r < 8; ++r) {
      int m = qbase + rb + r;
      if (m < NTOK) {
        float inv = 1.0f / lrow[r];
        size_t oi = ((size_t)(b * NTOK + m)) * DIM + h * DH;
        obuf[oi + cn]      = (_Float16)(o0[r] * inv);
        obuf[oi + 16 + cn] = (_Float16)(o1[r] * inv);
      }
    }
  }
}

// ---------------------------------------------------------------------------
// Elementwise / small kernels
// ---------------------------------------------------------------------------
__global__ __launch_bounds__(256) void layernorm_kernel(
    const float* __restrict__ x, const float* __restrict__ g,
    const float* __restrict__ bt, _Float16* __restrict__ y)
{
  __shared__ float red[256];
  int row = blockIdx.x, c = threadIdx.x;
  float v = x[(size_t)row * DIM + c];
  red[c] = v; __syncthreads();
#pragma unroll
  for (int s = 128; s > 0; s >>= 1) { if (c < s) red[c] += red[c + s]; __syncthreads(); }
  float mean = red[0] * (1.0f / DIM);
  __syncthreads();
  float d = v - mean;
  red[c] = d * d; __syncthreads();
#pragma unroll
  for (int s = 128; s > 0; s >>= 1) { if (c < s) red[c] += red[c + s]; __syncthreads(); }
  float var = red[0] * (1.0f / DIM);
  y[(size_t)row * DIM + c] = (_Float16)(d * rsqrtf(var + 1e-5f) * g[c] + bt[c]);
}

// mixed 2D RoPE (in-place on f16 q and k inside qkv buffer)
__global__ __launch_bounds__(256) void rope_kernel(
    _Float16* __restrict__ qkv, const float* __restrict__ fx,
    const float* __restrict__ fy)
{
  int idx = blockIdx.x * 256 + threadIdx.x;       // (b, n, h, f)
  if (idx >= BATCH * NTOK * HEADS * (DH / 2)) return;
  int f  = idx & 15;
  int h  = (idx >> 4) & 7;
  int bn = idx >> 7;
  int n  = bn % NTOK;
  int b  = bn / NTOK;
  float tx = (float)(n % PSZ), ty = (float)(n / PSZ);
  float ph = tx * fx[h * 16 + f] + ty * fy[h * 16 + f];
  float c = __cosf(ph), s = __sinf(ph);
  size_t base = ((size_t)(b * NTOK + n)) * (3 * DIM) + h * DH + 2 * f;
  float re = (float)qkv[base], im = (float)qkv[base + 1];
  qkv[base]     = (_Float16)(re * c - im * s);
  qkv[base + 1] = (_Float16)(re * s + im * c);
  re = (float)qkv[base + DIM]; im = (float)qkv[base + DIM + 1];
  qkv[base + DIM]     = (_Float16)(re * c - im * s);
  qkv[base + DIM + 1] = (_Float16)(re * s + im * c);
}

// convert f32 weight (K, N) -> f16 transposed (N, K) for async GEMM staging
__global__ __launch_bounds__(256) void wcvt_t_kernel(
    const float* __restrict__ w, _Float16* __restrict__ wt, int K, int N)
{
  int idx = blockIdx.x * 256 + threadIdx.x;
  if (idx >= K * N) return;
  int k = idx / N, n = idx % N;
  wt[(size_t)n * K + k] = (_Float16)w[idx];
}

// transpose conv weights (DIM, SPECT*9) -> (SPECT*9, DIM) for coalesced reads
__global__ __launch_bounds__(256) void wt_kernel(const float* __restrict__ w,
                                                 float* __restrict__ wtp)
{
  int i = blockIdx.x, c = threadIdx.x;   // i < 1800, c < 256
  wtp[(size_t)i * DIM + c] = w[(size_t)c * (SPECT * 9) + i];
}

// Conv2d(3x3, SAME) + BN (eval) + ReLU -> tok (B, N, DIM) f32
__global__ __launch_bounds__(256) void conv_kernel(
    const float* __restrict__ x, const float* __restrict__ wtp,
    const float* __restrict__ cb, const float* __restrict__ bg,
    const float* __restrict__ bb, const float* __restrict__ bm,
    const float* __restrict__ bv, float* __restrict__ tok)
{
  __shared__ float patch[SPECT * 9];
  int bn = blockIdx.x;
  int b  = bn / NTOK, n = bn % NTOK;
  int py0 = n / PSZ, px0 = n % PSZ;
  int tid = threadIdx.x;
  for (int e = tid; e < SPECT * 9; e += 256) {
    int ci = e / 9, kk = e % 9;
    int py = py0 + kk / 3 - 1, px = px0 + kk % 3 - 1;
    float v = 0.f;
    if (py >= 0 && py < PSZ && px >= 0 && px < PSZ)
      v = x[(((size_t)b * SPECT + ci) * PSZ + py) * PSZ + px];
    patch[e] = v;
  }
  __syncthreads();
  int co = tid;
  float acc = cb[co];
  for (int i = 0; i < SPECT * 9; ++i) acc += patch[i] * wtp[(size_t)i * DIM + co];
  acc = (acc - bm[co]) * bg[co] * rsqrtf(bv[co] + 1e-5f) + bb[co];
  acc = fmaxf(acc, 0.f);
  tok[(size_t)bn * DIM + co] = acc;
}

__global__ __launch_bounds__(256) void center_kernel(const float* __restrict__ tok,
                                                     float* __restrict__ centers)
{
  int b = blockIdx.x, c = threadIdx.x;
  centers[(size_t)b * DIM + c] = tok[((size_t)(b * NTOK + NTOK / 2)) * DIM + c];
}

__global__ __launch_bounds__(256) void head_kernel(
    const float* __restrict__ tok, const float* __restrict__ centers,
    const float* __restrict__ cw, const float* __restrict__ g,
    const float* __restrict__ bt, const float* __restrict__ W,
    const float* __restrict__ bias, float* __restrict__ out)
{
  __shared__ float nh[256];
  __shared__ float red[256];
  int b = blockIdx.x, c = threadIdx.x;
  float f = tok[((size_t)(b * NTOK + NTOK / 2)) * DIM + c];
#pragma unroll
  for (int i = 0; i < DEPTH; ++i)
    f += centers[((size_t)i * BATCH + b) * DIM + c] * cw[i];
  red[c] = f; __syncthreads();
#pragma unroll
  for (int s = 128; s > 0; s >>= 1) { if (c < s) red[c] += red[c + s]; __syncthreads(); }
  float mean = red[0] * (1.0f / DIM);
  __syncthreads();
  float d = f - mean;
  red[c] = d * d; __syncthreads();
#pragma unroll
  for (int s = 128; s > 0; s >>= 1) { if (c < s) red[c] += red[c + s]; __syncthreads(); }
  float var = red[0] * (1.0f / DIM);
  nh[c] = d * rsqrtf(var + 1e-5f) * g[c] + bt[c];
  __syncthreads();
  if (c < NCLS) {
    float acc = bias[c];
    for (int k = 0; k < DIM; ++k) acc += nh[k] * W[k * NCLS + c];
    out[b * NCLS + c] = acc;
  }
}

// ---------------------------------------------------------------------------
// Launch
// ---------------------------------------------------------------------------
extern "C" void kernel_launch(void* const* d_in, const int* in_sizes, int n_in,
                              void* d_out, int out_size, void* d_ws, size_t ws_size,
                              hipStream_t stream)
{
  const float* x        = (const float*)d_in[0];
  const float* conv_w   = (const float*)d_in[1];
  const float* conv_b   = (const float*)d_in[2];
  const float* bn_g     = (const float*)d_in[3];
  const float* bn_b     = (const float*)d_in[4];
  const float* bn_m     = (const float*)d_in[5];
  const float* bn_v     = (const float*)d_in[6];
  const float* freqs    = (const float*)d_in[7];
  const float* ln1_g    = (const float*)d_in[8];
  const float* ln1_b    = (const float*)d_in[9];
  const float* qkv_w    = (const float*)d_in[10];
  const float* out_w    = (const float*)d_in[11];
  const float* out_b    = (const float*)d_in[12];
  const float* ln2_g    = (const float*)d_in[13];
  const float* ln2_b    = (const float*)d_in[14];
  const float* mlp_w1   = (const float*)d_in[15];
  const float* mlp_b1   = (const float*)d_in[16];
  const float* mlp_w2   = (const float*)d_in[17];
  const float* mlp_b2   = (const float*)d_in[18];
  const float* cw       = (const float*)d_in[19];
  const float* cls_g    = (const float*)d_in[20];
  const float* cls_b    = (const float*)d_in[21];
  const float* cls_w    = (const float*)d_in[22];
  const float* cls_bias = (const float*)d_in[23];
  (void)in_sizes; (void)n_in; (void)out_size; (void)ws_size;

  char* ws = (char*)d_ws;
  size_t off = 0;
  auto allocB = [&](size_t bytes) {
    void* p = ws + off;
    off += (bytes + 255) & ~(size_t)255;
    return p;
  };
  float*     wtp     = (float*)    allocB((size_t)SPECT * 9 * DIM * 4);
  float*     tok     = (float*)    allocB((size_t)MROWS * DIM * 4);
  _Float16*  hbuf    = (_Float16*) allocB((size_t)MROWS * DIM * 2);
  _Float16*  obuf    = (_Float16*) allocB((size_t)MROWS * DIM * 2);
  _Float16*  big16   = (_Float16*) allocB((size_t)MROWS * MLPD * 2);
  float*     centers = (float*)    allocB((size_t)DEPTH * BATCH * DIM * 4);
  _Float16*  qkv_wh  = (_Float16*) allocB((size_t)DEPTH * DIM * 3 * DIM * 2);
  _Float16*  out_wh  = (_Float16*) allocB((size_t)DEPTH * DIM * DIM * 2);
  _Float16*  m1_wh   = (_Float16*) allocB((size_t)DEPTH * DIM * MLPD * 2);
  _Float16*  m2_wh   = (_Float16*) allocB((size_t)DEPTH * MLPD * DIM * 2);

  // One-time weight convert + transpose to (N, K) f16.
  for (int i = 0; i < DEPTH; ++i) {
    wcvt_t_kernel<<<(DIM * 3 * DIM) / 256, 256, 0, stream>>>(
        qkv_w + (size_t)i * DIM * 3 * DIM, qkv_wh + (size_t)i * 3 * DIM * DIM,
        DIM, 3 * DIM);
    wcvt_t_kernel<<<(DIM * DIM) / 256, 256, 0, stream>>>(
        out_w + (size_t)i * DIM * DIM, out_wh + (size_t)i * DIM * DIM, DIM, DIM);
    wcvt_t_kernel<<<(DIM * MLPD) / 256, 256, 0, stream>>>(
        mlp_w1 + (size_t)i * DIM * MLPD, m1_wh + (size_t)i * DIM * MLPD, DIM, MLPD);
    wcvt_t_kernel<<<(MLPD * DIM) / 256, 256, 0, stream>>>(
        mlp_w2 + (size_t)i * MLPD * DIM, m2_wh + (size_t)i * MLPD * DIM, MLPD, DIM);
  }

  wt_kernel<<<SPECT * 9, 256, 0, stream>>>(conv_w, wtp);
  conv_kernel<<<MROWS, 256, 0, stream>>>(x, wtp, conv_b, bn_g, bn_b, bn_m, bn_v, tok);

  for (int i = 0; i < DEPTH; ++i) {
    layernorm_kernel<<<MROWS, 256, 0, stream>>>(tok, ln1_g + i * DIM,
                                                ln1_b + i * DIM, hbuf);
    // QKV: f16 out, no bias/resid/act
    gemm_wmma_kernel<0, false, false, true>
        <<<dim3((3 * DIM) / BN, MROWS / BM), 256, 0, stream>>>(
        hbuf, qkv_wh + (size_t)i * 3 * DIM * DIM, nullptr, big16,
        nullptr, nullptr, MROWS, 3 * DIM, DIM);
    rope_kernel<<<(BATCH * NTOK * HEADS * (DH / 2) + 255) / 256, 256, 0, stream>>>(
        big16, freqs + i * 128, freqs + 512 + i * 128);
    attn_wmma_kernel<<<BATCH * HEADS, 256, 0, stream>>>(big16, obuf);
    // out-proj: f32 out + bias + residual(tok)
    gemm_wmma_kernel<0, true, true, false>
        <<<dim3(DIM / BN, MROWS / BM), 256, 0, stream>>>(
        obuf, out_wh + (size_t)i * DIM * DIM, tok, nullptr,
        out_b + i * DIM, tok, MROWS, DIM, DIM);
    layernorm_kernel<<<MROWS, 256, 0, stream>>>(tok, ln2_g + i * DIM,
                                                ln2_b + i * DIM, hbuf);
    // MLP1: f16 out + bias + exact GELU
    gemm_wmma_kernel<1, true, false, true>
        <<<dim3(MLPD / BN, MROWS / BM), 256, 0, stream>>>(
        hbuf, m1_wh + (size_t)i * DIM * MLPD, nullptr, big16,
        mlp_b1 + i * MLPD, nullptr, MROWS, MLPD, DIM);
    // MLP2: f32 out + bias + residual(tok)
    gemm_wmma_kernel<0, true, true, false>
        <<<dim3(DIM / BN, MROWS / BM), 256, 0, stream>>>(
        big16, m2_wh + (size_t)i * MLPD * DIM, tok, nullptr,
        mlp_b2 + i * DIM, tok, MROWS, DIM, MLPD);
    center_kernel<<<BATCH, 256, 0, stream>>>(tok, centers + (size_t)i * BATCH * DIM);
  }
  head_kernel<<<BATCH, 256, 0, stream>>>(tok, centers, cw, cls_g, cls_b, cls_w,
                                         cls_bias, (float*)d_out);
}
